// Qwen3MoeFusedExperts_21638045237561
// MI455X (gfx1250) — compile-verified, matrix operand order checked by reading, not run on hardware
//
#include <hip/hip_runtime.h>

// ---------------------------------------------------------------------------
// Qwen3 MoE fused experts for MI455X (gfx1250, wave32, WMMA bf16 path)
// Double-buffered LDS pipeline with gfx1250 async global->LDS copies for the
// activation tiles; weight tiles converted fp32->bf16 through VGPRs.
//
// Sizes (fixed by the reference):
#define E_   64
#define H_   2048
#define I_   768
#define T_   1024
#define K_   8
#define CAP_ 256
//
// GEMM tiling: block = 256 threads = 8 waves, block tile 128(M) x 64(N),
// K-chunk 64 staged in LDS (bf16), each wave owns a 32x32 output tile made of
// 2x2 v_wmma_f32_16x16x32_bf16 accumulators.
#define TM  128
#define TN  64
#define KC  64
#define LDK 80   // padded LDS row stride (bf16): 160B -> ds_load_b128 aligned, bank-spread
// ---------------------------------------------------------------------------

typedef __bf16 bf16;
typedef __attribute__((ext_vector_type(16))) __bf16 v16bf;
typedef __attribute__((ext_vector_type(8)))  __bf16 v8bf;
typedef __attribute__((ext_vector_type(8)))  float  v8f;
typedef __attribute__((ext_vector_type(4)))  int    v4i;

#if defined(__gfx1250__) && __has_builtin(__builtin_amdgcn_global_load_async_to_lds_b128)
#define ASYNC_LDS 1
typedef __attribute__((address_space(1))) v4i as1_v4i;   // global int4
typedef __attribute__((address_space(3))) v4i as3_v4i;   // LDS int4
#else
#define ASYNC_LDS 0
#endif

__device__ __forceinline__ void wait_async_lds() {
#if ASYNC_LDS
#if __has_builtin(__builtin_amdgcn_s_wait_asynccnt)
  __builtin_amdgcn_s_wait_asynccnt(0);
#else
  asm volatile("s_wait_asynccnt 0x0" ::: "memory");
#endif
#endif
}

#if ASYNC_LDS
// 64B (4 x b128) async copy, global -> LDS, no VGPR staging (ASYNCcnt-tracked)
__device__ __forceinline__ void async_copy64B(bf16* dst, const bf16* src) {
#pragma unroll
  for (int j = 0; j < 4; ++j)
    __builtin_amdgcn_global_load_async_to_lds_b128(
        (as1_v4i*)(src + 8 * j), (as3_v4i*)(dst + 8 * j), 0, 0);
}
#endif

__device__ __forceinline__ v8f wmma_bf16(v16bf a, v16bf b, v8f c) {
  // 8 args: (neg_a, A, neg_b, B, c_mod, C, reuse_a, reuse_b)
  return __builtin_amdgcn_wmma_f32_16x16x32_bf16(false, a, false, b, (short)0, c,
                                                 false, false);
}

// Fragment loader for a 16x32 (rows x K) bf16 tile stored row-major in LDS with
// stride LDK. Matches the ISA 16-bit operand layout:
//   lanes 0-15 : row=lane,    VGPR0-3 = K 0..7,  VGPR4-7 = K 16..23
//   lanes 16-31: row=lane-16, VGPR0-3 = K 8..15, VGPR4-7 = K 24..31
__device__ __forceinline__ v16bf load_frag16(const bf16* __restrict__ base, int lane) {
  const int row = lane & 15;
  const int kb  = (lane >> 4) << 3;           // 0 or 8
  const bf16* p = base + row * LDK + kb;
  v8bf lo = *(const v8bf*)(p);                // K kb .. kb+7     (16B)
  v8bf hi = *(const v8bf*)(p + 16);           // K kb+16 .. kb+23 (16B)
  return __builtin_shufflevector(lo, hi, 0, 1, 2, 3, 4, 5, 6, 7,
                                          8, 9, 10, 11, 12, 13, 14, 15);
}

__device__ __forceinline__ v8bf cvt8(float4 a, float4 b) {
  v8bf r;
  r[0] = (bf16)a.x; r[1] = (bf16)a.y; r[2] = (bf16)a.z; r[3] = (bf16)a.w;
  r[4] = (bf16)b.x; r[5] = (bf16)b.y; r[6] = (bf16)b.z; r[7] = (bf16)b.w;
  return r;
}

// ---------------------------------------------------------------------------
// Kernel 0: zero per-expert occupancy counters
__global__ void moe_zero_counts(int* __restrict__ counts) {
  if (threadIdx.x < E_) counts[threadIdx.x] = 0;
}

// ---------------------------------------------------------------------------
// Kernel 1: dispatch. One block per (token, k) pair: claim a slot in the
// expert's buffer and scatter the token row as bf16.
__global__ __launch_bounds__(256) void moe_dispatch(
    const float* __restrict__ hs, const int* __restrict__ sel,
    int* __restrict__ counts, int* __restrict__ posArr,
    bf16* __restrict__ bufA) {
  const int i = blockIdx.x;                   // flat (t*K + k)
  const int e = sel[i];
  __shared__ int spos;
  if (threadIdx.x == 0) {
    int p = atomicAdd(&counts[e], 1);
    posArr[i] = p;
    spos = p;
  }
  __syncthreads();
  const int p = spos;
  if (p >= CAP_) return;                      // drop (never at these sizes)
  const int t = i / K_;
  const float4* src = (const float4*)(hs + (size_t)t * H_) + threadIdx.x * 2;
  float4 a = src[0], b = src[1];
  *(v8bf*)(bufA + ((size_t)e * CAP_ + p) * H_ + threadIdx.x * 8) = cvt8(a, b);
}

// ---------------------------------------------------------------------------
// Kernel 2: gate+up grouped GEMM + SwiGLU -> bf16 hmid. Double-buffered.
//   A = bufA[e]   (CAP x H, bf16)  -> async global->LDS
//   Bg/Bu = gate/up weights (I x H, fp32, cvt->bf16 through VGPRs)
__global__ __launch_bounds__(256) void moe_gateup(
    const bf16* __restrict__ bufA, const float* __restrict__ gateW,
    const float* __restrict__ upW, const int* __restrict__ counts,
    bf16* __restrict__ hmid) {
  const int e = blockIdx.z;
  const int rowStart = blockIdx.y * TM;
  if (rowStart >= counts[e]) return;          // expert tile empty: skip
  const int colStart = blockIdx.x * TN;

  __shared__ bf16 sA[2][TM * LDK];
  __shared__ bf16 sBg[2][TN * LDK];
  __shared__ bf16 sBu[2][TN * LDK];

  const int tid = threadIdx.x;
  const int lane = tid & 31;
  const int wave = tid >> 5;
  const int wm = wave >> 1;                   // 0..3 (M waves)
  const int wn = wave & 1;                    // 0..1 (N waves)

  v8f accG[2][2] = {};
  v8f accU[2][2] = {};

  const int aRow = tid >> 1, aCol = (tid & 1) * 32;   // A: 2 thr/row, 32 bf16 each
  const int bRow = tid >> 2, bCol = (tid & 3) * 16;   // B: 4 thr/row, 16 f32 each

  const bf16*  aPtr = bufA + ((size_t)e * CAP_ + rowStart + aRow) * H_ + aCol;
  const float* gPtr = gateW + ((size_t)e * I_ + colStart + bRow) * H_ + bCol;
  const float* uPtr = upW   + ((size_t)e * I_ + colStart + bRow) * H_ + bCol;

  const int aOff = aRow * LDK + aCol;
  const int bOff = bRow * LDK + bCol;

  float4 bgReg[4], buReg[4];
#if !ASYNC_LDS
  uint4 aReg[4];
#endif

  // ---- prologue: stage k0 = 0 into buffer 0
#if ASYNC_LDS
  async_copy64B(&sA[0][aOff], aPtr);
#else
  {
    const uint4* g = (const uint4*)aPtr;
    uint4* d = (uint4*)&sA[0][aOff];
#pragma unroll
    for (int j = 0; j < 4; ++j) d[j] = g[j];
  }
#endif
  {
    const float4* g = (const float4*)gPtr;
    const float4* u = (const float4*)uPtr;
#pragma unroll
    for (int j = 0; j < 4; ++j) { bgReg[j] = g[j]; buReg[j] = u[j]; }
    v8bf* dg = (v8bf*)&sBg[0][bOff];
    dg[0] = cvt8(bgReg[0], bgReg[1]);
    dg[1] = cvt8(bgReg[2], bgReg[3]);
    v8bf* du = (v8bf*)&sBu[0][bOff];
    du[0] = cvt8(buReg[0], buReg[1]);
    du[1] = cvt8(buReg[2], buReg[3]);
  }
  wait_async_lds();
  __syncthreads();

  for (int k0 = 0; k0 < H_; k0 += KC) {
    const int cur = (k0 / KC) & 1;
    const int nxt = cur ^ 1;
    const bool hasNext = (k0 + KC) < H_;

    // ---- issue next-stage copies before compute (overlap with WMMA)
    if (hasNext) {
#if ASYNC_LDS
      async_copy64B(&sA[nxt][aOff], aPtr + KC);
#else
      const uint4* ga = (const uint4*)(aPtr + KC);
#pragma unroll
      for (int j = 0; j < 4; ++j) aReg[j] = ga[j];
#endif
      const float4* g = (const float4*)(gPtr + KC);
      const float4* u = (const float4*)(uPtr + KC);
#pragma unroll
      for (int j = 0; j < 4; ++j) { bgReg[j] = g[j]; buReg[j] = u[j]; }
      if (k0 + 2 * KC < H_) {                 // far prefetch into L2
        __builtin_prefetch(gPtr + 2 * KC, 0, 1);
        __builtin_prefetch(uPtr + 2 * KC, 0, 1);
      }
    }

    // ---- compute on current stage
#pragma unroll
    for (int kk = 0; kk < KC; kk += 32) {
      v16bf a0 = load_frag16(&sA[cur][(wm * 32 +  0) * LDK + kk], lane);
      v16bf a1 = load_frag16(&sA[cur][(wm * 32 + 16) * LDK + kk], lane);
      v16bf g0 = load_frag16(&sBg[cur][(wn * 32 +  0) * LDK + kk], lane);
      v16bf g1 = load_frag16(&sBg[cur][(wn * 32 + 16) * LDK + kk], lane);
      v16bf u0 = load_frag16(&sBu[cur][(wn * 32 +  0) * LDK + kk], lane);
      v16bf u1 = load_frag16(&sBu[cur][(wn * 32 + 16) * LDK + kk], lane);
      accG[0][0] = wmma_bf16(a0, g0, accG[0][0]);
      accG[0][1] = wmma_bf16(a0, g1, accG[0][1]);
      accG[1][0] = wmma_bf16(a1, g0, accG[1][0]);
      accG[1][1] = wmma_bf16(a1, g1, accG[1][1]);
      accU[0][0] = wmma_bf16(a0, u0, accU[0][0]);
      accU[0][1] = wmma_bf16(a0, u1, accU[0][1]);
      accU[1][0] = wmma_bf16(a1, u0, accU[1][0]);
      accU[1][1] = wmma_bf16(a1, u1, accU[1][1]);
    }

    // ---- finish next-stage staging, single barrier per iteration
    if (hasNext) {
#if !ASYNC_LDS
      uint4* da = (uint4*)&sA[nxt][aOff];
#pragma unroll
      for (int j = 0; j < 4; ++j) da[j] = aReg[j];
#endif
      v8bf* dg = (v8bf*)&sBg[nxt][bOff];
      dg[0] = cvt8(bgReg[0], bgReg[1]);
      dg[1] = cvt8(bgReg[2], bgReg[3]);
      v8bf* du = (v8bf*)&sBu[nxt][bOff];
      du[0] = cvt8(buReg[0], buReg[1]);
      du[1] = cvt8(buReg[2], buReg[3]);
      wait_async_lds();
    }
    __syncthreads();
    aPtr += KC; gPtr += KC; uPtr += KC;
  }

  // SwiGLU epilogue. C/D layout: VGPR r -> M = (lane>=16 ? 8 : 0)+r, N = lane&15
  const int rBase = rowStart + wm * 32 + ((lane >> 4) << 3);
  const int cBase = colStart + wn * 32 + (lane & 15);
#pragma unroll
  for (int ms = 0; ms < 2; ++ms)
#pragma unroll
    for (int ns = 0; ns < 2; ++ns)
#pragma unroll
      for (int r = 0; r < 8; ++r) {
        float gv = accG[ms][ns][r];
        float uv = accU[ms][ns][r];
        float s = gv / (1.0f + __expf(-gv)) * uv;
        int row = rBase + ms * 16 + r;
        int col = cBase + ns * 16;
        hmid[((size_t)e * CAP_ + row) * I_ + col] = (bf16)s;
      }
}

// ---------------------------------------------------------------------------
// Kernel 3: down grouped GEMM -> f32 y. Double-buffered.
//   A = hmid[e]  (CAP x I, bf16) -> async global->LDS
//   B = down weights (H x I, fp32 -> bf16 via VGPRs)
__global__ __launch_bounds__(256) void moe_down(
    const bf16* __restrict__ hmid, const float* __restrict__ downW,
    const int* __restrict__ counts, float* __restrict__ ybuf) {
  const int e = blockIdx.z;
  const int rowStart = blockIdx.y * TM;
  if (rowStart >= counts[e]) return;
  const int colStart = blockIdx.x * TN;

  __shared__ bf16 sA[2][TM * LDK];
  __shared__ bf16 sB[2][TN * LDK];

  const int tid = threadIdx.x;
  const int lane = tid & 31;
  const int wave = tid >> 5;
  const int wm = wave >> 1;
  const int wn = wave & 1;

  v8f acc[2][2] = {};

  const int aRow = tid >> 1, aCol = (tid & 1) * 32;
  const int bRow = tid >> 2, bCol = (tid & 3) * 16;

  const bf16*  aPtr = hmid + ((size_t)e * CAP_ + rowStart + aRow) * I_ + aCol;
  const float* wPtr = downW + ((size_t)e * H_ + colStart + bRow) * I_ + bCol;

  const int aOff = aRow * LDK + aCol;
  const int bOff = bRow * LDK + bCol;

  float4 bReg[4];
#if !ASYNC_LDS
  uint4 aReg[4];
#endif

  // prologue: stage k0 = 0 into buffer 0
#if ASYNC_LDS
  async_copy64B(&sA[0][aOff], aPtr);
#else
  {
    const uint4* g = (const uint4*)aPtr;
    uint4* d = (uint4*)&sA[0][aOff];
#pragma unroll
    for (int j = 0; j < 4; ++j) d[j] = g[j];
  }
#endif
  {
    const float4* g = (const float4*)wPtr;
#pragma unroll
    for (int j = 0; j < 4; ++j) bReg[j] = g[j];
    v8bf* d = (v8bf*)&sB[0][bOff];
    d[0] = cvt8(bReg[0], bReg[1]);
    d[1] = cvt8(bReg[2], bReg[3]);
  }
  wait_async_lds();
  __syncthreads();

  for (int k0 = 0; k0 < I_; k0 += KC) {
    const int cur = (k0 / KC) & 1;
    const int nxt = cur ^ 1;
    const bool hasNext = (k0 + KC) < I_;

    if (hasNext) {
#if ASYNC_LDS
      async_copy64B(&sA[nxt][aOff], aPtr + KC);
#else
      const uint4* ga = (const uint4*)(aPtr + KC);
#pragma unroll
      for (int j = 0; j < 4; ++j) aReg[j] = ga[j];
#endif
      const float4* g = (const float4*)(wPtr + KC);
#pragma unroll
      for (int j = 0; j < 4; ++j) bReg[j] = g[j];
      if (k0 + 2 * KC < I_)
        __builtin_prefetch(wPtr + 2 * KC, 0, 1);
    }

#pragma unroll
    for (int kk = 0; kk < KC; kk += 32) {
      v16bf a0 = load_frag16(&sA[cur][(wm * 32 +  0) * LDK + kk], lane);
      v16bf a1 = load_frag16(&sA[cur][(wm * 32 + 16) * LDK + kk], lane);
      v16bf b0 = load_frag16(&sB[cur][(wn * 32 +  0) * LDK + kk], lane);
      v16bf b1 = load_frag16(&sB[cur][(wn * 32 + 16) * LDK + kk], lane);
      acc[0][0] = wmma_bf16(a0, b0, acc[0][0]);
      acc[0][1] = wmma_bf16(a0, b1, acc[0][1]);
      acc[1][0] = wmma_bf16(a1, b0, acc[1][0]);
      acc[1][1] = wmma_bf16(a1, b1, acc[1][1]);
    }

    if (hasNext) {
#if !ASYNC_LDS
      uint4* da = (uint4*)&sA[nxt][aOff];
#pragma unroll
      for (int j = 0; j < 4; ++j) da[j] = aReg[j];
#endif
      v8bf* d = (v8bf*)&sB[nxt][bOff];
      d[0] = cvt8(bReg[0], bReg[1]);
      d[1] = cvt8(bReg[2], bReg[3]);
      wait_async_lds();
    }
    __syncthreads();
    aPtr += KC; wPtr += KC;
  }

  const int rBase = rowStart + wm * 32 + ((lane >> 4) << 3);
  const int cBase = colStart + wn * 32 + (lane & 15);
#pragma unroll
  for (int ms = 0; ms < 2; ++ms)
#pragma unroll
    for (int ns = 0; ns < 2; ++ns)
#pragma unroll
      for (int r = 0; r < 8; ++r) {
        int row = rBase + ms * 16 + r;
        int col = cBase + ns * 16;
        ybuf[((size_t)e * CAP_ + row) * H_ + col] = acc[ms][ns][r];
      }
}

// ---------------------------------------------------------------------------
// Kernel 4: combine. One block per token; gather its K expert rows, weight-sum.
__global__ __launch_bounds__(256) void moe_combine(
    const float* __restrict__ ybuf, const float* __restrict__ rw,
    const int* __restrict__ sel, const int* __restrict__ posArr,
    float* __restrict__ out) {
  const int t = blockIdx.x;
  const int c0 = threadIdx.x * 8;
  float4 acc0 = {0.f, 0.f, 0.f, 0.f};
  float4 acc1 = {0.f, 0.f, 0.f, 0.f};
#pragma unroll
  for (int kk = 0; kk < K_; ++kk) {
    const int i = t * K_ + kk;
    const int e = sel[i];
    const int p = posArr[i];
    const float w = rw[i];
    if (p < CAP_) {
      const float4* src = (const float4*)(ybuf + ((size_t)e * CAP_ + p) * H_ + c0);
      float4 a = src[0], b = src[1];
      acc0.x += w * a.x; acc0.y += w * a.y; acc0.z += w * a.z; acc0.w += w * a.w;
      acc1.x += w * b.x; acc1.y += w * b.y; acc1.z += w * b.z; acc1.w += w * b.w;
    }
  }
  float4* dst = (float4*)(out + (size_t)t * H_ + c0);
  dst[0] = acc0;
  dst[1] = acc1;
}

// ---------------------------------------------------------------------------
// Workspace layout (bytes), ~227 MB total:
//   [0,        256)        int counts[E]
//   [256,      33024)      int posArr[T*K]
//   [33024,    +67108864)  bf16 bufA [E][CAP][H]
//   [...,      +25165824)  bf16 hmid [E][CAP][I]
//   [...,      +134217728) f32  y    [E][CAP][H]
#define OFF_POS   256
#define OFF_BUFA  33024
#define OFF_HMID  (OFF_BUFA + (size_t)E_ * CAP_ * H_ * 2)
#define OFF_Y     (OFF_HMID + (size_t)E_ * CAP_ * I_ * 2)

extern "C" void kernel_launch(void* const* d_in, const int* in_sizes, int n_in,
                              void* d_out, int out_size, void* d_ws, size_t ws_size,
                              hipStream_t stream) {
  const float* hs  = (const float*)d_in[0];   // (T, H)
  const float* rw  = (const float*)d_in[1];   // (T, K)
  const int*   sel = (const int*)d_in[2];     // (T, K)
  const float* gp  = (const float*)d_in[3];   // (E, I, H)
  const float* up  = (const float*)d_in[4];   // (E, I, H)
  const float* dp  = (const float*)d_in[5];   // (E, H, I)
  float* out = (float*)d_out;

  char* ws = (char*)d_ws;
  int*   counts = (int*)ws;
  int*   posArr = (int*)(ws + OFF_POS);
  bf16*  bufA   = (bf16*)(ws + OFF_BUFA);
  bf16*  hmid   = (bf16*)(ws + OFF_HMID);
  float* ybuf   = (float*)(ws + OFF_Y);

  moe_zero_counts<<<1, 64, 0, stream>>>(counts);
  moe_dispatch<<<T_ * K_, 256, 0, stream>>>(hs, sel, counts, posArr, bufA);
  moe_gateup<<<dim3(I_ / TN, CAP_ / TM, E_), 256, 0, stream>>>(bufA, gp, up, counts, hmid);
  moe_down<<<dim3(H_ / TN, CAP_ / TM, E_), 256, 0, stream>>>(hmid, dp, counts, ybuf);
  moe_combine<<<T_, 256, 0, stream>>>(ybuf, rw, sel, posArr, out);
}
